// MultiHeadAttention_21165598835165
// MI455X (gfx1250) — matrix operand order repaired and checked by reading, not compiled, not run on hardware
//
#include <hip/hip_runtime.h>

// ---------------------------------------------------------------------------
// MI455X (gfx1250) multi-head attention forward.
// B=2, S=2048, D=1024, H=16, hd=64.  bf16 WMMA (f32 accum) everywhere.
// ---------------------------------------------------------------------------

#define S_LEN   2048
#define D_MOD   1024
#define N_HEAD  16
#define HDIM    64
#define BATCH   2
#define M_ROWS  (BATCH * S_LEN)   // 4096

typedef unsigned short u16s;
typedef __bf16  v16bf  __attribute__((ext_vector_type(16)));
typedef float   v8f    __attribute__((ext_vector_type(8)));
typedef unsigned int uint4v __attribute__((ext_vector_type(4)));
typedef unsigned short ush4 __attribute__((ext_vector_type(4)));
typedef float   f4v    __attribute__((ext_vector_type(4)));

union Frag {            // one WMMA A or B operand (16x32 / 32x16 bf16)
    v16bf v;
    uint4v q[2];
};

__device__ __forceinline__ u16s f2bf(float f) {
    unsigned u = __float_as_uint(f);
    unsigned r = (u + 0x7FFFu + ((u >> 16) & 1u)) >> 16;   // RNE
    return (u16s)r;
}

__device__ __forceinline__ v8f wmma_bf16(v16bf a, v16bf b, v8f c) {
    return __builtin_amdgcn_wmma_f32_16x16x32_bf16(
        /*neg_a=*/false, a, /*neg_b=*/false, b,
        /*c_mod=*/(short)0, c, /*reuse_a=*/false, /*reuse_b=*/false);
}

__device__ __forceinline__ ush4 cvt4(const float* p) {
    f4v f = *(const f4v*)p;
    ush4 r;
    r.x = f2bf(f.x); r.y = f2bf(f.y); r.z = f2bf(f.z); r.w = f2bf(f.w);
    return r;
}
__device__ __forceinline__ ush4 cvt4(const u16s* p) {
    return *(const ush4*)p;
}

__device__ __forceinline__ float rmax16(float v) {
    v = fmaxf(v, __shfl_xor(v, 1, 32));
    v = fmaxf(v, __shfl_xor(v, 2, 32));
    v = fmaxf(v, __shfl_xor(v, 4, 32));
    v = fmaxf(v, __shfl_xor(v, 8, 32));
    return v;
}
__device__ __forceinline__ float rsum16(float v) {
    v += __shfl_xor(v, 1, 32);
    v += __shfl_xor(v, 2, 32);
    v += __shfl_xor(v, 4, 32);
    v += __shfl_xor(v, 8, 32);
    return v;
}

// ---------------------------------------------------------------------------
// GEMM: Y[M,N] = X[M,K] * W[N,K]^T + bias, K=N=1024, M=4096.
// MODE 0: bf16 out, head-split [B,H,S,hd], *scale folded in (for Q/K)
// MODE 1: bf16 out, head-split transposed [B,H,hd,S]   (for V)
// MODE 2: f32 out, flat [M,N]                           (final projection)
// Block: 256 thr (8 waves), tile 128x64, k-tile 32; each wave owns 32x32 C
// (2 A-frags x 2 B-frags -> 4 WMMAs per k-step). LDS row stride 40 halves.
// ---------------------------------------------------------------------------
template <typename TX, int MODE>
__global__ __launch_bounds__(256) void gemm_wmma_kernel(
    const TX* __restrict__ X, const float* __restrict__ W,
    const float* __restrict__ bias, void* __restrict__ out, float scale)
{
    constexpr int K = D_MOD;
    constexpr int N = D_MOD;
    constexpr int LSTR = 40;                 // padded LDS row stride (halves)

    __shared__ alignas(16) u16s As[128 * LSTR];   // 10240 B
    __shared__ alignas(16) u16s Bs[64 * LSTR];    //  5120 B

    const int t    = threadIdx.x;
    const int w    = t >> 5;
    const int lane = t & 31;
    const int nl   = lane & 15;
    const int half = lane >> 4;
    const int wr   = w & 3;                  // 0..3 -> m offset (32 rows each)
    const int wc   = w >> 2;                 // 0..1 -> n offset (32 cols each)
    const int m0   = blockIdx.y * 128;
    const int n0   = blockIdx.x * 64;

    v8f c00 = {}, c01 = {}, c10 = {}, c11 = {};

    for (int kt = 0; kt < K; kt += 32) {
        __syncthreads();
#pragma unroll
        for (int p = 0; p < 4; ++p) {        // A tile: 128x32 f32 -> bf16
            int idx = (t + p * 256) * 4;     // 0..4095
            int row = idx >> 5;
            int col = idx & 31;
            *(ush4*)&As[row * LSTR + col] = cvt4(X + (size_t)(m0 + row) * K + kt + col);
            if (kt + 32 < K)                 // global_prefetch_b8 next k-tile
                __builtin_prefetch(X + (size_t)(m0 + row) * K + kt + 32 + col, 0, 1);
        }
#pragma unroll
        for (int p = 0; p < 2; ++p) {        // B tile: 64x32
            int idx = (t + p * 256) * 4;
            int row = idx >> 5;
            int col = idx & 31;
            *(ush4*)&Bs[row * LSTR + col] = cvt4(W + (size_t)(n0 + row) * K + kt + col);
            if (kt + 32 < K)
                __builtin_prefetch(W + (size_t)(n0 + row) * K + kt + 32 + col, 0, 1);
        }
        __syncthreads();

        // A fragments: rows wr*32 + {0,16} .. +15
        Frag a0, a1;
        const u16s* ar0 = As + (wr * 32 + nl) * LSTR;
        const u16s* ar1 = ar0 + 16 * LSTR;
        a0.q[0] = *(const uint4v*)(ar0 + half * 8);
        a0.q[1] = *(const uint4v*)(ar0 + 16 + half * 8);
        a1.q[0] = *(const uint4v*)(ar1 + half * 8);
        a1.q[1] = *(const uint4v*)(ar1 + 16 + half * 8);

        // B fragments: cols wc*32 + {0,16} .. +15
        Frag b0, b1;
        const u16s* br0 = Bs + (wc * 32 + nl) * LSTR;
        const u16s* br1 = br0 + 16 * LSTR;
        b0.q[0] = *(const uint4v*)(br0 + half * 8);
        b0.q[1] = *(const uint4v*)(br0 + 16 + half * 8);
        b1.q[0] = *(const uint4v*)(br1 + half * 8);
        b1.q[1] = *(const uint4v*)(br1 + 16 + half * 8);

        c00 = wmma_bf16(a0.v, b0.v, c00);
        c01 = wmma_bf16(a0.v, b1.v, c01);
        c10 = wmma_bf16(a1.v, b0.v, c10);
        c11 = wmma_bf16(a1.v, b1.v, c11);
    }

    const int gn0 = n0 + wc * 32 + nl;
    const int gn1 = gn0 + 16;
    const float bias0 = bias[gn0];
    const float bias1 = bias[gn1];

#pragma unroll
    for (int i = 0; i < 2; ++i) {
        const v8f& ci0 = i ? c10 : c00;
        const v8f& ci1 = i ? c11 : c01;
#pragma unroll
        for (int r = 0; r < 8; ++r) {
            const int gm = m0 + wr * 32 + i * 16 + r + 8 * half;  // C layout row
            float v0 = (ci0[r] + bias0) * scale;
            float v1 = (ci1[r] + bias1) * scale;
            if constexpr (MODE == 2) {
                float* O = (float*)out;
                O[(size_t)gm * N + gn0] = v0;
                O[(size_t)gm * N + gn1] = v1;
            } else {
                u16s* O = (u16s*)out;
                const int bI = gm >> 11;                   // / S_LEN
                const int s  = gm & (S_LEN - 1);
                const int h0 = gn0 >> 6, d0 = gn0 & 63;
                const int h1 = gn1 >> 6, d1 = gn1 & 63;
                if constexpr (MODE == 0) {                 // [B,H,S,hd]
                    O[(((size_t)bI * N_HEAD + h0) * S_LEN + s) * HDIM + d0] = f2bf(v0);
                    O[(((size_t)bI * N_HEAD + h1) * S_LEN + s) * HDIM + d1] = f2bf(v1);
                } else {                                   // [B,H,hd,S]
                    O[(((size_t)bI * N_HEAD + h0) * HDIM + d0) * S_LEN + s] = f2bf(v0);
                    O[(((size_t)bI * N_HEAD + h1) * HDIM + d1) * S_LEN + s] = f2bf(v1);
                }
            }
        }
    }
}

// ---------------------------------------------------------------------------
// Flash attention: 1 wave = 16 query rows; loop over keys in steps of 64
// (16 WMMAs per step: 8 score + 8 P*V). Q pre-scaled by 1/sqrt(hd).
// O accumulated as 4 f32 C-frags (16x64). Block 128 thr (4 waves).
// Grid (S/64, B*H).
// ---------------------------------------------------------------------------
__global__ __launch_bounds__(128) void attn_flash_kernel(
    const u16s* __restrict__ Q, const u16s* __restrict__ Kb,
    const u16s* __restrict__ Vt, u16s* __restrict__ AO)
{
    constexpr int LP = 72;                        // padded P-tile row stride
    __shared__ alignas(16) u16s p_lds[4][16 * LP];   // 9216 B

    const int w    = threadIdx.x >> 5;
    const int lane = threadIdx.x & 31;
    const int nl   = lane & 15;
    const int half = lane >> 4;
    const int bh   = blockIdx.y;                  // b*16 + h
    const int b    = bh >> 4;
    const int h    = bh & 15;
    const int qbase = blockIdx.x * 64 + w * 16;

    const u16s* Qp = Q  + ((size_t)bh * S_LEN + qbase) * HDIM;
    const u16s* Kp = Kb + (size_t)bh * S_LEN * HDIM;
    const u16s* Vp = Vt + (size_t)bh * HDIM * S_LEN;

    // Q A-fragments (dims 0..31 and 32..63), kept in registers all along.
    Frag aq0, aq1;
    const u16s* qrow = Qp + nl * HDIM;
    aq0.q[0] = *(const uint4v*)(qrow + half * 8);
    aq0.q[1] = *(const uint4v*)(qrow + 16 + half * 8);
    aq1.q[0] = *(const uint4v*)(qrow + 32 + half * 8);
    aq1.q[1] = *(const uint4v*)(qrow + 48 + half * 8);

    v8f o[4]; o[0] = v8f{}; o[1] = v8f{}; o[2] = v8f{}; o[3] = v8f{};
    float mrow[8], lrow[8];
#pragma unroll
    for (int r = 0; r < 8; ++r) { mrow[r] = -1e30f; lrow[r] = 0.0f; }

    for (int kb = 0; kb < S_LEN; kb += 64) {
        // ---- scores for 64 keys: four 16x16 tiles, contraction over hd=64
        v8f s[4];
#pragma unroll
        for (int q4 = 0; q4 < 4; ++q4) {
            const u16s* kp = Kp + (size_t)(kb + q4 * 16 + nl) * HDIM;
            Frag bk;
            bk.q[0] = *(const uint4v*)(kp + half * 8);
            bk.q[1] = *(const uint4v*)(kp + 16 + half * 8);
            v8f acc = {};
            acc = wmma_bf16(aq0.v, bk.v, acc);
            bk.q[0] = *(const uint4v*)(kp + 32 + half * 8);
            bk.q[1] = *(const uint4v*)(kp + 48 + half * 8);
            acc = wmma_bf16(aq1.v, bk.v, acc);
            s[q4] = acc;
        }

        // ---- online softmax (per-row stats; rows = r + 8*half)
        float alpha[8];
#pragma unroll
        for (int r = 0; r < 8; ++r) {
            float mx = fmaxf(fmaxf(s[0][r], s[1][r]), fmaxf(s[2][r], s[3][r]));
            mx = rmax16(mx);
            float mnew = fmaxf(mrow[r], mx);
            alpha[r] = __expf(mrow[r] - mnew);
            mrow[r] = mnew;
            float p0 = __expf(s[0][r] - mnew);
            float p1 = __expf(s[1][r] - mnew);
            float p2 = __expf(s[2][r] - mnew);
            float p3 = __expf(s[3][r] - mnew);
            s[0][r] = p0; s[1][r] = p1; s[2][r] = p2; s[3][r] = p3;
            lrow[r] = lrow[r] * alpha[r] + rsum16((p0 + p1) + (p2 + p3));
        }
#pragma unroll
        for (int tfr = 0; tfr < 4; ++tfr)
#pragma unroll
            for (int r = 0; r < 8; ++r) o[tfr][r] *= alpha[r];

        // ---- P tile (16x64) through wave-private LDS: C-layout -> A-layout
        u16s* pl = &p_lds[w][0];
#pragma unroll
        for (int q4 = 0; q4 < 4; ++q4)
#pragma unroll
            for (int r = 0; r < 8; ++r)
                pl[(r + 8 * half) * LP + q4 * 16 + nl] = f2bf(s[q4][r]);
        asm volatile("s_wait_dscnt 0" ::: "memory");   // cross-lane LDS RAW

        Frag ap0, ap1;                                 // keys kb+0..31 / +32..63
        const u16s* prow = pl + nl * LP;
        ap0.q[0] = *(const uint4v*)(prow + half * 8);
        ap0.q[1] = *(const uint4v*)(prow + 16 + half * 8);
        ap1.q[0] = *(const uint4v*)(prow + 32 + half * 8);
        ap1.q[1] = *(const uint4v*)(prow + 48 + half * 8);

        // ---- O += P * V  (V transposed: rows = dim, contiguous in key)
#pragma unroll
        for (int tfr = 0; tfr < 4; ++tfr) {
            const u16s* vrow = Vp + (size_t)(tfr * 16 + nl) * S_LEN + kb;
            Frag bv;
            bv.q[0] = *(const uint4v*)(vrow + half * 8);
            bv.q[1] = *(const uint4v*)(vrow + 16 + half * 8);
            o[tfr] = wmma_bf16(ap0.v, bv.v, o[tfr]);
            bv.q[0] = *(const uint4v*)(vrow + 32 + half * 8);
            bv.q[1] = *(const uint4v*)(vrow + 48 + half * 8);
            o[tfr] = wmma_bf16(ap1.v, bv.v, o[tfr]);
        }
        asm volatile("s_wait_dscnt 0" ::: "memory");   // reads done before rewrite
    }

    // ---- normalize and store merged-head bf16 output [B,S,D]
    float inv[8];
#pragma unroll
    for (int r = 0; r < 8; ++r) inv[r] = 1.0f / lrow[r];
#pragma unroll
    for (int tfr = 0; tfr < 4; ++tfr) {
#pragma unroll
        for (int r = 0; r < 8; ++r) {
            const int s = qbase + r + 8 * half;
            const int d = tfr * 16 + nl;
            AO[((size_t)b * S_LEN + s) * D_MOD + h * HDIM + d] = f2bf(o[tfr][r] * inv[r]);
        }
    }
}

// ---------------------------------------------------------------------------
extern "C" void kernel_launch(void* const* d_in, const int* in_sizes, int n_in,
                              void* d_out, int out_size, void* d_ws, size_t ws_size,
                              hipStream_t stream) {
    const float* query = (const float*)d_in[0];
    const float* key   = (const float*)d_in[1];
    const float* value = (const float*)d_in[2];
    const float* Wq = (const float*)d_in[3];
    const float* bq = (const float*)d_in[4];
    const float* Wk = (const float*)d_in[5];
    const float* bk = (const float*)d_in[6];
    const float* Wv = (const float*)d_in[7];
    const float* bv = (const float*)d_in[8];
    const float* Wo = (const float*)d_in[9];
    const float* bo = (const float*)d_in[10];
    float* out = (float*)d_out;

    // workspace: 4 x (4096*1024) bf16 = 32 MB
    u16s* Qbf  = (u16s*)d_ws;
    u16s* Kbf  = Qbf  + (size_t)M_ROWS * D_MOD;
    u16s* Vtbf = Kbf  + (size_t)M_ROWS * D_MOD;
    u16s* AObf = Vtbf + (size_t)M_ROWS * D_MOD;

    const dim3 gg(D_MOD / 64, M_ROWS / 128);  // (16, 32)
    const dim3 gb(256);
    const float qscale = 0.125f;              // 1/sqrt(hd)

    gemm_wmma_kernel<float, 0><<<gg, gb, 0, stream>>>(query, Wq, bq, Qbf, qscale);
    gemm_wmma_kernel<float, 0><<<gg, gb, 0, stream>>>(key,   Wk, bk, Kbf, 1.0f);
    gemm_wmma_kernel<float, 1><<<gg, gb, 0, stream>>>(value, Wv, bv, Vtbf, 1.0f);

    attn_flash_kernel<<<dim3(S_LEN / 64, BATCH * N_HEAD), 128, 0, stream>>>(
        Qbf, Kbf, Vtbf, AObf);

    gemm_wmma_kernel<u16s, 2><<<gg, gb, 0, stream>>>(AObf, Wo, bo, out, 1.0f);
}
